// EncoderBiLSTMMaxPool_74174085202236
// MI455X (gfx1250) — compile-verified, matrix operand order checked by reading, not run on hardware
//
#include <hip/hip_runtime.h>
#include <hip/hip_bf16.h>
#include <stdint.h>

// ---------------------------------------------------------------------------
// EncoderBiLSTMMaxPool for MI455X (gfx1250, wave32, WMMA)
//   B=S=128, H=512, HH=256, L=2, gates=1024 per dir, 2048 both dirs
// ---------------------------------------------------------------------------
#define B_   128
#define S_   128
#define H_   512
#define HH_  256
#define G4_  1024            // 4*HH (one direction's gates)
#define NC_  2048            // both directions' gates
#define T_   (B_ * S_)       // 16384 (sample,time) pairs

typedef __attribute__((ext_vector_type(16))) __bf16 v16bf;
typedef __attribute__((ext_vector_type(8)))  float  v8f;

__device__ __forceinline__ unsigned short f2bf(float x) {       // RNE f32->bf16
  unsigned u = __builtin_bit_cast(unsigned, x);
  u += 0x7fffu + ((u >> 16) & 1u);
  return (unsigned short)(u >> 16);
}
__device__ __forceinline__ float sigm(float x) { return 1.0f / (1.0f + __expf(-x)); }

// Load a 16x32 bf16 fragment (A-layout per ISA 7.12.2) from a row-major bf16
// matrix: lane<16 holds row=lane with K {0..7,16..23}; lane>=16 holds the same
// rows with K {8..15,24..31}.  B (32x16 = rows^T) uses the mirrored layout, so
// passing the *row-major weight matrix* (rows = output columns) through the
// same loader yields the B fragment.
__device__ __forceinline__ v16bf load_frag(const unsigned short* __restrict__ base,
                                           int row0, int k0, int ld) {
  int lane = threadIdx.x & 31;
  int r    = lane & 15;
  int kk   = (lane & 16) ? 8 : 0;
  const unsigned short* p = base + (row0 + r) * ld + k0 + kk;   // 16B aligned
  union { uint4 u[2]; v16bf v; } f;
  f.u[0] = *(const uint4*)(p);        // K = kk .. kk+7
  f.u[1] = *(const uint4*)(p + 16);   // K = kk+16 .. kk+23
  return f.v;
}

// ---------------- prep kernels ----------------
__global__ __launch_bounds__(256) void prep_bias(const float* __restrict__ bih,
                                                 const float* __restrict__ bhh,
                                                 float* __restrict__ b) {
  int i = blockIdx.x * 256 + threadIdx.x;        // < 4096 = L*NC_
  b[i] = bih[i] + bhh[i];
}

__global__ __launch_bounds__(256) void prep_wih_bf16(const float* __restrict__ W,
                                                     unsigned short* __restrict__ Wb) {
  int i = blockIdx.x * 256 + threadIdx.x;        // < 2*2048*512
  Wb[i] = f2bf(W[i]);
}

// WhhT[(l*2+d)][k][j] = Whh[l][d][j][k]  (f32, coalesced reads in the GEMV)
__global__ __launch_bounds__(256) void prep_whhT(const float* __restrict__ Whh,
                                                 float* __restrict__ Wt) {
  int i  = blockIdx.x * 256 + threadIdx.x;       // < 4*256*1024
  int j  = i & (G4_ - 1);
  int k  = (i >> 10) & (HH_ - 1);
  int ld = i >> 18;                              // l*2+d
  Wt[i] = Whh[(ld * G4_ + j) * HH_ + k];
}

// ---------------- embedding gather:  emb = E_rt[rt]+E_re[re]+E_rm[rm] (bf16)
__global__ __launch_bounds__(256) void embed_sum(
    const int* __restrict__ rt, const int* __restrict__ re, const int* __restrict__ rm,
    const float* __restrict__ Et, const float* __restrict__ Ee, const float* __restrict__ Em,
    unsigned short* __restrict__ emb) {
  int t  = blockIdx.x;                 // 0..16383
  int o1 = rt[t] * H_, o2 = re[t] * H_, o3 = rm[t] * H_;
  for (int h = threadIdx.x; h < H_; h += 256)
    emb[t * H_ + h] = f2bf(Et[o1 + h] + Ee[o2 + h] + Em[o3 + h]);
}

// ---------------- big input-transform GEMM (bf16 WMMA, f32 accum) ----------
//   G[t][n] = sum_k A[t][k] * W[n][k] + bias[n]      A: T_ x 512, W: 2048 x 512
//   16x64 register tile per wave: 1 A fragment feeds 4 WMMAs per K-step
//   (1.25 fragment loads per WMMA instead of 2).  64 WMMAs per wave.
__global__ __launch_bounds__(256) void gemm_gates_bf16(
    const unsigned short* __restrict__ A, const unsigned short* __restrict__ W,
    const float* __restrict__ bias, float* __restrict__ G) {
  int wave = threadIdx.x >> 5;
  int lane = threadIdx.x & 31;
  int tile = blockIdx.x * 8 + wave;    // 1024 Mtiles * 32 Ngroups = 32768
  int ng   = tile & 31;                // N-group of 64 columns
  int mt   = tile >> 5;
  int m0 = mt << 4, n0 = ng << 6;
  v8f acc0 = {0.f,0.f,0.f,0.f,0.f,0.f,0.f,0.f};
  v8f acc1 = acc0, acc2 = acc0, acc3 = acc0;
  #pragma unroll
  for (int k0 = 0; k0 < 512; k0 += 32) {
    v16bf a  = load_frag(A, m0, k0, 512);
    v16bf b0 = load_frag(W, n0,      k0, 512);
    v16bf b1 = load_frag(W, n0 + 16, k0, 512);
    v16bf b2 = load_frag(W, n0 + 32, k0, 512);
    v16bf b3 = load_frag(W, n0 + 48, k0, 512);
    acc0 = __builtin_amdgcn_wmma_f32_16x16x32_bf16(false, a, false, b0, (short)0, acc0, false, false);
    acc1 = __builtin_amdgcn_wmma_f32_16x16x32_bf16(false, a, false, b1, (short)0, acc1, false, false);
    acc2 = __builtin_amdgcn_wmma_f32_16x16x32_bf16(false, a, false, b2, (short)0, acc2, false, false);
    acc3 = __builtin_amdgcn_wmma_f32_16x16x32_bf16(false, a, false, b3, (short)0, acc3, false, false);
  }
  // C/D layout: VGPR r, lanes 0-15 -> M=r,N=lane ; lanes 16-31 -> M=8+r,N=lane-16
  int nl = lane & 15;
  int mb = m0 + ((lane & 16) ? 8 : 0);
  v8f accs[4] = {acc0, acc1, acc2, acc3};
  #pragma unroll
  for (int i = 0; i < 4; ++i) {
    int   n  = n0 + (i << 4) + nl;
    float bv = bias[n];
    #pragma unroll
    for (int r = 0; r < 8; ++r)
      G[(mb + r) * NC_ + n] = accs[i][r] + bv;
  }
}

// ---------------- recurrent LSTM layer --------------------------------------
// 8 blocks = 4 reset-separated sample-chains x 2 directions; 1024 threads:
// thread j owns gate j; h (256) broadcast from LDS; WhhT streamed from L2.
__global__ __launch_bounds__(1024) void lstm_layer_k(
    const float* __restrict__ G,        // T_ x 2048 pre-activations (+bias)
    const float* __restrict__ WhhT,     // this layer: [2][256][1024]
    const float* __restrict__ h0, const float* __restrict__ c0,
    int layer,
    float* __restrict__ outF,           // layer 1: d_out (bilstm_outs), else null
    unsigned short* __restrict__ outB)  // layer 0: bf16 feed to next GEMM, else null
{
  __shared__ float h_s[HH_];
  __shared__ float g_s[G4_];
  int chain = blockIdx.x >> 1;
  int d     = blockIdx.x & 1;
  int j     = threadIdx.x;
  const float* Wt = WhhT + d * (HH_ * G4_);
  float cst = 0.f;
  if (j < HH_) {
    float hv = 0.f;
    if (chain == 0) {                   // only first chain uses provided h0/c0
      hv  = h0[(2 * layer + d) * HH_ + j];
      cst = c0[(2 * layer + d) * HH_ + j];
    }
    h_s[j] = hv;
  }
  __syncthreads();
  for (int e = chain * 32; e < chain * 32 + 32; ++e) {      // samples in chain
    for (int ss = 0; ss < S_; ++ss) {
      int s = d ? (S_ - 1 - ss) : ss;                       // reverse for bwd
      int t = e * S_ + s;
      float acc = G[t * NC_ + d * G4_ + j];
      #pragma unroll 8
      for (int k = 0; k < HH_; ++k) {                       // 256-long GEMV
        if ((k & 63) == 0)                                  // L2 prefetch ahead
          __builtin_prefetch(&Wt[(k + 64 < HH_ ? k + 64 : k) * G4_ + j], 0, 0);
        acc = __builtin_fmaf(h_s[k], Wt[k * G4_ + j], acc);
      }
      g_s[j] = acc;
      __syncthreads();
      if (j < HH_) {                                        // gate order i,f,g,o
        float ig = sigm(g_s[j]);
        float fg = sigm(g_s[HH_ + j]);
        float gg = tanhf(g_s[2 * HH_ + j]);
        float og = sigm(g_s[3 * HH_ + j]);
        cst = fg * cst + ig * gg;
        float hv = og * tanhf(cst);
        h_s[j] = hv;
        int oi = t * H_ + d * HH_ + j;                      // concat [yf|yb]
        if (outF) outF[oi] = hv;
        if (outB) outB[oi] = f2bf(hv);
      }
      __syncthreads();
    }
  }
}

// ---------------- max over sample axis --------------------------------------
__global__ __launch_bounds__(256) void maxpool_b(const float* __restrict__ Y,
                                                 float* __restrict__ out) {
  int i = blockIdx.x * 256 + threadIdx.x;        // < S_*H_ = 65536
  float m = -3.402823466e38f;
  for (int b = 0; b < B_; ++b) m = fmaxf(m, Y[b * (S_ * H_) + i]);
  out[i] = m;
}

// ---------------------------------------------------------------------------
extern "C" void kernel_launch(void* const* d_in, const int* in_sizes, int n_in,
                              void* d_out, int out_size, void* d_ws, size_t ws_size,
                              hipStream_t stream) {
  (void)in_sizes; (void)n_in; (void)out_size; (void)ws_size;
  const int*   rt  = (const int*)d_in[0];
  const int*   re  = (const int*)d_in[1];
  const int*   rm  = (const int*)d_in[2];
  const float* h0  = (const float*)d_in[3];
  const float* c0  = (const float*)d_in[4];
  const float* Ert = (const float*)d_in[5];
  const float* Ere = (const float*)d_in[6];
  const float* Erm = (const float*)d_in[7];
  const float* Wih = (const float*)d_in[8];
  const float* Whh = (const float*)d_in[9];
  const float* bih = (const float*)d_in[10];
  const float* bhh = (const float*)d_in[11];
  float* out = (float*)d_out;

  // workspace carve-up (~176 MB peak; gates buffer reused between layers)
  char*  ws  = (char*)d_ws;
  size_t off = 0;
  auto take = [&](size_t bytes) { size_t o = off; off += (bytes + 255) & ~size_t(255); return o; };
  unsigned short* emb  = (unsigned short*)(ws + take((size_t)T_ * H_ * 2));       // 16 MB
  unsigned short* wihb = (unsigned short*)(ws + take((size_t)2 * NC_ * H_ * 2));  //  4 MB
  float*          whhT = (float*)         (ws + take((size_t)4 * HH_ * G4_ * 4)); //  4 MB
  float*          bias = (float*)         (ws + take((size_t)2 * NC_ * 4));       // 16 KB
  float*          G    = (float*)         (ws + take((size_t)T_ * NC_ * 4));      // 128 MB
  unsigned short* y1   = (unsigned short*)(ws + take((size_t)T_ * H_ * 2));       // 16 MB

  // prep
  prep_bias    <<<16,    256, 0, stream>>>(bih, bhh, bias);
  prep_wih_bf16<<<8192,  256, 0, stream>>>(Wih, wihb);
  prep_whhT    <<<4096,  256, 0, stream>>>(Whh, whhT);
  // embedding gather (bf16)
  embed_sum    <<<T_,    256, 0, stream>>>(rt, re, rm, Ert, Ere, Erm, emb);
  // layer 0: input GEMM (both dirs) + 8-chain recurrence -> y1 (bf16)
  gemm_gates_bf16<<<4096,  256, 0, stream>>>(emb, wihb, bias, G);
  lstm_layer_k   <<<8,    1024, 0, stream>>>(G, whhT, h0, c0, 0, nullptr, y1);
  // layer 1: input GEMM on y1 + recurrence -> d_out (bilstm_outs)
  gemm_gates_bf16<<<4096,  256, 0, stream>>>(y1, wihb + (size_t)NC_ * H_, bias + NC_, G);
  lstm_layer_k   <<<8,    1024, 0, stream>>>(G, whhT + (size_t)2 * HH_ * G4_, h0, c0, 1, out, nullptr);
  // max over sample axis -> second output region
  maxpool_b      <<<256,   256, 0, stream>>>(out, out + (size_t)B_ * S_ * H_);
}
